// VRNN_197568496029
// MI455X (gfx1250) — matrix operand order repaired and checked by reading, not compiled, not run on hardware
//
#include <hip/hip_runtime.h>

// ---------------- dims ----------------
#define BQ   256   // batch (GEMM M)
#define TT   512   // time steps
#define XX   64    // input dim
#define PXD  256   // phi_x / phi_z dim
#define HD   512   // hidden dim
#define LD   256   // latent dim
#define G3   1536  // 3*H gate dim
#define NBLK 32
#define NTHR 256
#define NWAVE ((NBLK * NTHR) / 32)

typedef __bf16 v16bf __attribute__((ext_vector_type(16)));
typedef __bf16 v8bf  __attribute__((ext_vector_type(8)));
typedef float  v8f   __attribute__((ext_vector_type(8)));

// ---------------- scalar converters ----------------
__device__ __forceinline__ __bf16 f2bf(float f) {
  union { float f; unsigned u; } a; a.f = f;
  unsigned r = a.u + 0x7FFFu + ((a.u >> 16) & 1u);   // round-to-nearest-even
  union { unsigned short s; __bf16 b; } c; c.s = (unsigned short)(r >> 16);
  return c.b;
}
__device__ __forceinline__ float bf2f(__bf16 b) {
  union { unsigned short s; __bf16 b; } c; c.b = b;
  union { unsigned u; float f; } a; a.u = ((unsigned)c.s) << 16;
  return a.f;
}
__device__ __forceinline__ float sigm(float x) { return 1.0f / (1.0f + __expf(-x)); }

// ---------------- WMMA ----------------
__device__ __forceinline__ v8f wmma_bf(v16bf a, v16bf b, v8f c) {
  return __builtin_amdgcn_wmma_f32_16x16x32_bf16(false, a, false, b, (short)0, c, false, false);
}

// A fragment (16x32 bf16): lane<16 -> row m0+lane, K = kb*32 + {0..7, 16..23}
//                          lane>=16 -> row m0+lane-16, K = kb*32 + {8..15, 24..31}
__device__ __forceinline__ v16bf load_a_bf(const __bf16* A, int lda, int m0, int kb, int lane) {
  const __bf16* p = A + (size_t)(m0 + (lane & 15)) * lda + kb * 32 + ((lane >> 4) << 3);
  v8bf lo = *(const v8bf*)p;
  v8bf hi = *(const v8bf*)(p + 16);
  return __builtin_shufflevector(lo, hi, 0,1,2,3,4,5,6,7,8,9,10,11,12,13,14,15);
}
// A fragment sourced from fp32 memory (converted on the fly)
__device__ __forceinline__ v16bf load_a_f32(const float* A, int lda, int m0, int kb, int lane) {
  const float* p = A + (size_t)(m0 + (lane & 15)) * lda + kb * 32 + ((lane >> 4) << 3);
  v16bf a;
#pragma unroll
  for (int i = 0; i < 8; ++i) { a[i] = f2bf(p[i]); a[i + 8] = f2bf(p[16 + i]); }
  return a;
}
// B fragment (32x16 bf16) from pre-transposed weights WT[N][K]:
// lane holds K values of its column: K = kb*32 + (lane>=16 ? {8..15, 24..31} : {0..7, 16..23})
__device__ __forceinline__ v16bf load_b(const __bf16* WT, int ldk, int n0, int kb, int lane) {
  const __bf16* p = WT + (size_t)(n0 + (lane & 15)) * ldk + kb * 32 + ((lane >> 4) << 3);
  v8bf lo = *(const v8bf*)p;
  v8bf hi = *(const v8bf*)(p + 16);
  return __builtin_shufflevector(lo, hi, 0,1,2,3,4,5,6,7,8,9,10,11,12,13,14,15);
}

// C tile epilogues. f32 16x16 C layout: VGPR r -> row m0 + r + (lane>=16 ? 8:0), col n0 + lane%16
__device__ __forceinline__ void store_bf(__bf16* O, int ldo, int m0, int n0, v8f c,
                                         const float* bias, bool relu, int lane) {
  int col = n0 + (lane & 15);
  int r0  = m0 + ((lane >> 4) << 3);
  float bv = bias[col];
#pragma unroll
  for (int r = 0; r < 8; ++r) {
    float v = c[r] + bv;
    if (relu) v = fmaxf(v, 0.0f);
    O[(size_t)(r0 + r) * ldo + col] = f2bf(v);
  }
}
__device__ __forceinline__ void store_f32(float* O, int ldo, int m0, int n0, v8f c,
                                          const float* bias, int lane) {
  int col = n0 + (lane & 15);
  int r0  = m0 + ((lane >> 4) << 3);
  float bv = bias[col];
#pragma unroll
  for (int r = 0; r < 8; ++r) O[(size_t)(r0 + r) * ldo + col] = c[r] + bv;
}
// store into d_out section laid out [B][T][ldo]
__device__ __forceinline__ void store_out(float* O, int t, int ldo, int m0, int n0, v8f c,
                                          const float* bias, int lane) {
  int col = n0 + (lane & 15);
  int r0  = m0 + ((lane >> 4) << 3);
  float bv = bias[col];
#pragma unroll
  for (int r = 0; r < 8; ++r)
    O[((size_t)(r0 + r) * TT + t) * ldo + col] = c[r] + bv;
}

// ---------------- grid-wide barrier ----------------
__device__ __forceinline__ void grid_sync(unsigned* bar) {
  __syncthreads();
  if (threadIdx.x == 0) {
    __threadfence();
    volatile unsigned* vgen = bar + 1;
    unsigned g = *vgen;
    unsigned old = atomicAdd(bar, 1u);
    if (old == (unsigned)(NBLK - 1)) {
      *(volatile unsigned*)bar = 0u;
      __threadfence();
      atomicAdd(bar + 1, 1u);
    } else {
      while (*vgen == g) __builtin_amdgcn_s_sleep(1);
    }
    __threadfence();
  }
  __syncthreads();
}

// transpose-convert W[K][N] (fp32) -> WT[N][K] (bf16)
__device__ void tconv(const float* W, __bf16* WT, int K, int N, int tid, int nt) {
  int total = K * N;
  for (int i = tid; i < total; i += nt) {
    int n = i / K, k = i - n * K;
    WT[i] = f2bf(W[(size_t)k * N + n]);
  }
}

// ---------------- params ----------------
struct VP {
  const float *x, *eps;
  const float *W_phix, *b_phix, *W_enc, *b_enc, *W_mu, *b_mu, *W_lv, *b_lv;
  const float *W_pp, *b_pp, *W_mup, *b_mup, *W_lvp, *b_lvp, *W_phiz, *b_phiz;
  const float *W_dec, *b_dec, *W_ih, *W_hh, *b_ih, *b_hh;
  float *o_out, *o_mu, *o_lv, *o_mup, *o_lvp;
  float *hf0, *hf1, *gh;
  __bf16 *hbf0, *hbf1, *phiX, *zp, *enc, *zb, *phiz;
  __bf16 *WT_phix, *WT_enc, *WT_mu, *WT_lv, *WT_pp, *WT_mup, *WT_lvp, *WT_phiz, *WT_ih, *WT_hh;
  unsigned* bar;
};

// ---------------- one timestep (inlined twice; all pointers kernarg-direct
// so every access infers to the GLOBAL address space, no flat_load) ----------
__device__ __forceinline__ void vrnn_step(const VP& p, int t,
                                          const float* hc, const __bf16* hb,
                                          float* hfn, __bf16* hbn,
                                          int tid, int wid, int lane) {
  // ---- P1: phi_x = relu(x W + b), zp = relu(h W_pp + b), gh = h W_hh + b_hh ----
  for (int i = wid; i < 2048; i += NWAVE) {
    if (i < 256) {
      int m0 = (i & 15) << 4, n0 = (i >> 4) << 4;
      v8f acc = {};
#pragma unroll
      for (int kb = 0; kb < 2; ++kb) {
        v16bf a = load_a_f32(p.x + (size_t)t * XX, TT * XX, m0, kb, lane);
        v16bf b = load_b(p.WT_phix, XX, n0, kb, lane);
        acc = wmma_bf(a, b, acc);
      }
      store_bf(p.phiX, PXD, m0, n0, acc, p.b_phix, true, lane);
    } else if (i < 512) {
      int j = i - 256, m0 = (j & 15) << 4, n0 = (j >> 4) << 4;
      v8f acc = {};
#pragma unroll
      for (int kb = 0; kb < 16; ++kb) {
        v16bf a = load_a_bf(hb, HD, m0, kb, lane);
        v16bf b = load_b(p.WT_pp, HD, n0, kb, lane);
        acc = wmma_bf(a, b, acc);
      }
      store_bf(p.zp, 256, m0, n0, acc, p.b_pp, true, lane);
    } else {
      int j = i - 512, m0 = (j & 15) << 4, n0 = (j >> 4) << 4;
      v8f acc = {};
#pragma unroll
      for (int kb = 0; kb < 16; ++kb) {
        v16bf a = load_a_bf(hb, HD, m0, kb, lane);
        v16bf b = load_b(p.WT_hh, HD, n0, kb, lane);
        acc = wmma_bf(a, b, acc);
      }
      store_f32(p.gh, G3, m0, n0, acc, p.b_hh, lane);
    }
  }
  grid_sync(p.bar);

  // ---- P2: enc = relu([phi_x,h] W_enc + b), mup = zp W_mup + b, lvp = zp W_lvp + b ----
  for (int i = wid; i < 768; i += NWAVE) {
    if (i < 256) {
      int m0 = (i & 15) << 4, n0 = (i >> 4) << 4;
      v8f acc = {};
#pragma unroll
      for (int kb = 0; kb < 24; ++kb) {
        v16bf a = (kb < 8) ? load_a_bf(p.phiX, PXD, m0, kb, lane)
                           : load_a_bf(hb, HD, m0, kb - 8, lane);
        v16bf b = load_b(p.WT_enc, PXD + HD, n0, kb, lane);
        acc = wmma_bf(a, b, acc);
      }
      store_bf(p.enc, 256, m0, n0, acc, p.b_enc, true, lane);
    } else if (i < 512) {
      int j = i - 256, m0 = (j & 15) << 4, n0 = (j >> 4) << 4;
      v8f acc = {};
#pragma unroll
      for (int kb = 0; kb < 8; ++kb) {
        v16bf a = load_a_bf(p.zp, 256, m0, kb, lane);
        v16bf b = load_b(p.WT_mup, 256, n0, kb, lane);
        acc = wmma_bf(a, b, acc);
      }
      store_out(p.o_mup, t, LD, m0, n0, acc, p.b_mup, lane);
    } else {
      int j = i - 512, m0 = (j & 15) << 4, n0 = (j >> 4) << 4;
      v8f acc = {};
#pragma unroll
      for (int kb = 0; kb < 8; ++kb) {
        v16bf a = load_a_bf(p.zp, 256, m0, kb, lane);
        v16bf b = load_b(p.WT_lvp, 256, n0, kb, lane);
        acc = wmma_bf(a, b, acc);
      }
      store_out(p.o_lvp, t, LD, m0, n0, acc, p.b_lvp, lane);
    }
  }
  grid_sync(p.bar);

  // ---- P3: mu, lv, z = mu + eps * exp(0.5 lv) ----
  for (int i = wid; i < 256; i += NWAVE) {
    int m0 = (i & 15) << 4, n0 = (i >> 4) << 4;
    v8f amu = {}, alv = {};
#pragma unroll
    for (int kb = 0; kb < 8; ++kb) {
      v16bf a   = load_a_bf(p.enc, 256, m0, kb, lane);
      v16bf bmu = load_b(p.WT_mu, 256, n0, kb, lane);
      v16bf blv = load_b(p.WT_lv, 256, n0, kb, lane);
      amu = wmma_bf(a, bmu, amu);
      alv = wmma_bf(a, blv, alv);
    }
    int col = n0 + (lane & 15);
    int r0  = m0 + ((lane >> 4) << 3);
    float bmu = p.b_mu[col], blv = p.b_lv[col];
#pragma unroll
    for (int r = 0; r < 8; ++r) {
      int row = r0 + r;
      float mu = amu[r] + bmu;
      float lv = alv[r] + blv;
      size_t oidx = ((size_t)row * TT + t) * LD + col;
      p.o_mu[oidx] = mu;
      p.o_lv[oidx] = lv;
      float z = mu + p.eps[oidx] * __expf(0.5f * lv);
      p.zb[(size_t)row * LD + col] = f2bf(z);
    }
  }
  grid_sync(p.bar);

  // ---- P4: phi_z = relu(z W_phiz + b) ----
  for (int i = wid; i < 256; i += NWAVE) {
    int m0 = (i & 15) << 4, n0 = (i >> 4) << 4;
    v8f acc = {};
#pragma unroll
    for (int kb = 0; kb < 8; ++kb) {
      v16bf a = load_a_bf(p.zb, LD, m0, kb, lane);
      v16bf b = load_b(p.WT_phiz, LD, n0, kb, lane);
      acc = wmma_bf(a, b, acc);
    }
    store_bf(p.phiz, PXD, m0, n0, acc, p.b_phiz, true, lane);
  }
  grid_sync(p.bar);

  // ---- P5: GRU gates (gi = [phi_x,phi_z] W_ih + b), h update, decoder out ----
  for (int i = wid; i < 512; i += NWAVE) {
    int m0 = (i & 15) << 4;
    int n0 = (i >> 4) << 4;           // hidden-unit column block, 0..496
    v8f ar = {}, az = {}, an = {};
#pragma unroll
    for (int kb = 0; kb < 16; ++kb) {
      v16bf a = (kb < 8) ? load_a_bf(p.phiX, PXD, m0, kb, lane)
                         : load_a_bf(p.phiz, PXD, m0, kb - 8, lane);
      v16bf br = load_b(p.WT_ih, HD, n0,          kb, lane);
      v16bf bz = load_b(p.WT_ih, HD, HD + n0,     kb, lane);
      v16bf bn = load_b(p.WT_ih, HD, 2 * HD + n0, kb, lane);
      ar = wmma_bf(a, br, ar);
      az = wmma_bf(a, bz, az);
      an = wmma_bf(a, bn, an);
    }
    int j  = n0 + (lane & 15);
    int r0 = m0 + ((lane >> 4) << 3);
    float bir = p.b_ih[j], biz = p.b_ih[HD + j], bin = p.b_ih[2 * HD + j];
#pragma unroll
    for (int r = 0; r < 8; ++r) {
      int row = r0 + r;
      const float* ghr = p.gh + (size_t)row * G3;
      float rg  = sigm(ar[r] + bir + ghr[j]);
      float zg  = sigm(az[r] + biz + ghr[HD + j]);
      float nn  = tanhf(an[r] + bin + rg * ghr[2 * HD + j]);
      float hp  = hc[(size_t)row * HD + j];
      float hn2 = (1.0f - zg) * nn + zg * hp;
      hfn[(size_t)row * HD + j] = hn2;
      hbn[(size_t)row * HD + j] = f2bf(hn2);
    }
  }
  // decoder: out = relu([phi_z, h_old] W_dec + b_dec), OUT=2 -> per-thread dots
  if (tid < BQ * 2) {
    int row = tid >> 1, col = tid & 1;
    float acc = p.b_dec[col];
    const __bf16* pz = p.phiz + (size_t)row * PXD;
    for (int k = 0; k < PXD; ++k) acc += bf2f(pz[k]) * p.W_dec[k * 2 + col];
    const float* hr = hc + (size_t)row * HD;
    for (int k = 0; k < HD; ++k) acc += hr[k] * p.W_dec[(PXD + k) * 2 + col];
    p.o_out[((size_t)row * TT + t) * 2 + col] = fmaxf(acc, 0.0f);
  }
  grid_sync(p.bar);
}

// ---------------- persistent VRNN kernel ----------------
__global__ __launch_bounds__(NTHR, 1) void vrnn_kernel(VP p) {
  const int tid  = blockIdx.x * NTHR + threadIdx.x;
  const int NT   = NBLK * NTHR;
  const int lane = threadIdx.x & 31;
  const int wid  = tid >> 5;

  // ---- init: zero h, build bf16 transposed weights ----
  for (int i = tid; i < BQ * HD; i += NT) {
    p.hf0[i] = 0.0f; p.hf1[i] = 0.0f;
    p.hbf0[i] = f2bf(0.0f); p.hbf1[i] = f2bf(0.0f);
  }
  tconv(p.W_phix, p.WT_phix, XX,        PXD, tid, NT);
  tconv(p.W_enc,  p.WT_enc,  PXD + HD,  256, tid, NT);
  tconv(p.W_mu,   p.WT_mu,   256,       LD,  tid, NT);
  tconv(p.W_lv,   p.WT_lv,   256,       LD,  tid, NT);
  tconv(p.W_pp,   p.WT_pp,   HD,        256, tid, NT);
  tconv(p.W_mup,  p.WT_mup,  256,       LD,  tid, NT);
  tconv(p.W_lvp,  p.WT_lvp,  256,       LD,  tid, NT);
  tconv(p.W_phiz, p.WT_phiz, LD,        PXD, tid, NT);
  tconv(p.W_ih,   p.WT_ih,   PXD + PXD, G3,  tid, NT);
  tconv(p.W_hh,   p.WT_hh,   HD,        G3,  tid, NT);
  grid_sync(p.bar);

#pragma unroll 1
  for (int t = 0; t < TT; t += 2) {
    // even step: h in buf0 -> buf1 ; odd step: buf1 -> buf0
    vrnn_step(p, t,     p.hf0, p.hbf0, p.hf1, p.hbf1, tid, wid, lane);
    vrnn_step(p, t + 1, p.hf1, p.hbf1, p.hf0, p.hbf0, tid, wid, lane);
  }
}

// ---------------- host launch ----------------
extern "C" void kernel_launch(void* const* d_in, const int* in_sizes, int n_in,
                              void* d_out, int out_size, void* d_ws, size_t ws_size,
                              hipStream_t stream) {
  (void)in_sizes; (void)n_in; (void)out_size; (void)ws_size;
  VP p;
  p.x      = (const float*)d_in[0];
  p.eps    = (const float*)d_in[1];
  p.W_phix = (const float*)d_in[2];  p.b_phix = (const float*)d_in[3];
  p.W_enc  = (const float*)d_in[4];  p.b_enc  = (const float*)d_in[5];
  p.W_mu   = (const float*)d_in[6];  p.b_mu   = (const float*)d_in[7];
  p.W_lv   = (const float*)d_in[8];  p.b_lv   = (const float*)d_in[9];
  p.W_pp   = (const float*)d_in[10]; p.b_pp   = (const float*)d_in[11];
  p.W_mup  = (const float*)d_in[12]; p.b_mup  = (const float*)d_in[13];
  p.W_lvp  = (const float*)d_in[14]; p.b_lvp  = (const float*)d_in[15];
  p.W_phiz = (const float*)d_in[16]; p.b_phiz = (const float*)d_in[17];
  p.W_dec  = (const float*)d_in[18]; p.b_dec  = (const float*)d_in[19];
  p.W_ih   = (const float*)d_in[20]; p.W_hh   = (const float*)d_in[21];
  p.b_ih   = (const float*)d_in[22]; p.b_hh   = (const float*)d_in[23];

  float* out = (float*)d_out;
  p.o_out = out;
  p.o_mu  = out + (size_t)BQ * TT * 2;
  p.o_lv  = p.o_mu  + (size_t)BQ * TT * LD;
  p.o_mup = p.o_lv  + (size_t)BQ * TT * LD;
  p.o_lvp = p.o_mup + (size_t)BQ * TT * LD;

  char* w = (char*)d_ws;
  auto alloc = [&](size_t bytes) -> char* {
    char* r = w;
    w += (bytes + 255) & ~(size_t)255;
    return r;
  };
  p.hf0  = (float*)alloc((size_t)BQ * HD * 4);
  p.hf1  = (float*)alloc((size_t)BQ * HD * 4);
  p.hbf0 = (__bf16*)alloc((size_t)BQ * HD * 2);
  p.hbf1 = (__bf16*)alloc((size_t)BQ * HD * 2);
  p.phiX = (__bf16*)alloc((size_t)BQ * PXD * 2);
  p.zp   = (__bf16*)alloc((size_t)BQ * 256 * 2);
  p.enc  = (__bf16*)alloc((size_t)BQ * 256 * 2);
  p.zb   = (__bf16*)alloc((size_t)BQ * LD * 2);
  p.phiz = (__bf16*)alloc((size_t)BQ * PXD * 2);
  p.gh   = (float*)alloc((size_t)BQ * G3 * 4);
  p.WT_phix = (__bf16*)alloc((size_t)XX * PXD * 2);
  p.WT_enc  = (__bf16*)alloc((size_t)(PXD + HD) * 256 * 2);
  p.WT_mu   = (__bf16*)alloc((size_t)256 * LD * 2);
  p.WT_lv   = (__bf16*)alloc((size_t)256 * LD * 2);
  p.WT_pp   = (__bf16*)alloc((size_t)HD * 256 * 2);
  p.WT_mup  = (__bf16*)alloc((size_t)256 * LD * 2);
  p.WT_lvp  = (__bf16*)alloc((size_t)256 * LD * 2);
  p.WT_phiz = (__bf16*)alloc((size_t)LD * PXD * 2);
  p.WT_ih   = (__bf16*)alloc((size_t)(PXD + PXD) * G3 * 2);
  p.WT_hh   = (__bf16*)alloc((size_t)HD * G3 * 2);
  p.bar     = (unsigned*)alloc(256);

  hipMemsetAsync(p.bar, 0, 256, stream);
  vrnn_kernel<<<NBLK, NTHR, 0, stream>>>(p);
}
